// MultiHeadAttention_721554505804
// MI455X (gfx1250) — matrix operand order repaired and checked by reading, not compiled, not run on hardware
//
#include <hip/hip_runtime.h>
#include <hip/hip_bf16.h>
#include <stdint.h>
#include <stddef.h>

#define SEQ 4096
#define EMB 1024
#define NH  16
#define DK  64

typedef __bf16 bf16_t;
typedef __bf16 v8bf  __attribute__((ext_vector_type(8)));
typedef __bf16 v16bf __attribute__((ext_vector_type(16)));
typedef float  v8f   __attribute__((ext_vector_type(8)));

static __device__ __forceinline__ bf16_t f2bf(float f) {
  uint32_t u = __builtin_bit_cast(uint32_t, f);
  u += 0x7FFFu + ((u >> 16) & 1u);           // round-to-nearest-even
  uint16_t h = (uint16_t)(u >> 16);
  return __builtin_bit_cast(bf16_t, h);
}

// A-operand (16xK32 tile, row-major source, row = lane&15, K chunks at (lane>>4)*8 and +16)
static __device__ __forceinline__ v16bf load_a_nt(const bf16_t* p, int ld) {
  const int lane = (int)(threadIdx.x & 31u);
  const bf16_t* q = p + (size_t)(lane & 15) * ld + ((lane >> 4) << 3);
  v8bf lo = *(const v8bf*)(q);
  v8bf hi = *(const v8bf*)(q + 16);
  return __builtin_shufflevector(lo, hi, 0,1,2,3,4,5,6,7,8,9,10,11,12,13,14,15);
}

// B-operand (K32x16 tile given as row-major Bt[n][k]; col = lane&15, K chunks at
// (lane>>4)*8 and +16, same order as the A-operand).
static __device__ __forceinline__ v16bf load_b_nt(const bf16_t* p, int ld) {
  const int lane = (int)(threadIdx.x & 31u);
  const bf16_t* q = p + (size_t)(lane & 15) * ld + ((lane >> 4) << 3);
  v8bf lo = *(const v8bf*)(q);
  v8bf hi = *(const v8bf*)(q + 16);
  return __builtin_shufflevector(lo, hi, 0,1,2,3,4,5,6,7,8,9,10,11,12,13,14,15);
}

__global__ void cvt_f32_bf16(const float* __restrict__ in, bf16_t* __restrict__ out, int n) {
  int i = (int)(blockIdx.x * blockDim.x + threadIdx.x);
  if (i < n) out[i] = f2bf(in[i]);
}

// C[M,N] = scale * (A[M,K] * B[N,K]^T + bias[N])
// Wave tile 64x64 (4x4 WMMA accumulators), 8 waves/block -> 128(M) x 256(N) block tile.
// MODE 0: f32 row-major out; MODE 1: bf16 row-major out; MODE 2: bf16 transposed out[n*M+m]
template <int MODE>
__global__ __launch_bounds__(256)
void gemm_bf16_wmma(const bf16_t* __restrict__ A, const bf16_t* __restrict__ B,
                    const float* __restrict__ bias, void* __restrict__ out,
                    int M, int N, int K, float scale) {
  const int lane  = (int)(threadIdx.x & 31u);
  const int wave  = (int)(threadIdx.x >> 5);
  const int waveM = wave & 1, waveN = wave >> 1;   // 2 x 4 waves
  const int rowBase = (int)blockIdx.y * 128 + waveM * 64;
  const int colBase = (int)blockIdx.x * 256 + waveN * 64;
  const int hi = lane >> 4, lo16 = lane & 15;

  v8f acc[4][4];
#pragma unroll
  for (int mt = 0; mt < 4; ++mt)
#pragma unroll
    for (int nt = 0; nt < 4; ++nt)
      acc[mt][nt] = (v8f){0.f,0.f,0.f,0.f,0.f,0.f,0.f,0.f};

  for (int k = 0; k < K; k += 32) {
    if (k + 32 < K) {
      __builtin_prefetch(A + (size_t)rowBase * K + k + 32, 0, 1);
      __builtin_prefetch(B + (size_t)colBase * K + k + 32, 0, 1);
    }
    v16bf a[4], b[4];
#pragma unroll
    for (int mt = 0; mt < 4; ++mt)
      a[mt] = load_a_nt(A + (size_t)(rowBase + mt * 16) * K + k, K);
#pragma unroll
    for (int nt = 0; nt < 4; ++nt)
      b[nt] = load_b_nt(B + (size_t)(colBase + nt * 16) * K + k, K);
#pragma unroll
    for (int mt = 0; mt < 4; ++mt)
#pragma unroll
      for (int nt = 0; nt < 4; ++nt)
        acc[mt][nt] = __builtin_amdgcn_wmma_f32_16x16x32_bf16(
            false, a[mt], false, b[nt], (short)0, acc[mt][nt], false, false);
  }

#pragma unroll
  for (int nt = 0; nt < 4; ++nt) {
    const float bv = bias[colBase + nt * 16 + lo16];
#pragma unroll
    for (int mt = 0; mt < 4; ++mt) {
#pragma unroll
      for (int r = 0; r < 8; ++r) {
        const int m = rowBase + mt * 16 + r + 8 * hi;
        const int n = colBase + nt * 16 + lo16;
        const float v = (acc[mt][nt][r] + bv) * scale;
        if (MODE == 0)      ((float*)out)[(size_t)m * N + n]  = v;
        else if (MODE == 1) ((bf16_t*)out)[(size_t)m * N + n] = f2bf(v);
        else                ((bf16_t*)out)[(size_t)n * M + m] = f2bf(v);
      }
    }
  }
}

// Causal flash attention. One wave = one (head, 32 q-rows) block; k processed 64 columns/iter.
// Qb,Kb: [S][E] bf16 (head h at column h*64, Q pre-scaled by 1/sqrt(dk));
// Vt: [E][S] bf16 (row h*64+d = V column); ctx out: [S][E] bf16.
__global__ __launch_bounds__(128)
void attn_causal_wmma(const bf16_t* __restrict__ Qb, const bf16_t* __restrict__ Kb,
                      const bf16_t* __restrict__ Vt, bf16_t* __restrict__ ctx) {
  __shared__ __align__(16) bf16_t plds[4][32 * 64];   // 16 KB: per-wave 32x64 P tile
  const int lane = (int)(threadIdx.x & 31u);
  const int wave = (int)(threadIdx.x >> 5);
  const int gid  = (int)blockIdx.x * 4 + wave;     // 0..2047
  const int head = gid >> 7;                       // 128 q-blocks per head
  const int q0   = (gid & 127) * 32;
  const int hi = lane >> 4, lo16 = lane & 15;
  const int hoff = head * DK;

  // Q A-operands: 2 M-tiles x 2 K-chunks (d_k = 64)
  v16bf qa[2][2];
#pragma unroll
  for (int mt = 0; mt < 2; ++mt)
#pragma unroll
    for (int kc = 0; kc < 2; ++kc)
      qa[mt][kc] = load_a_nt(Qb + (size_t)(q0 + mt * 16) * EMB + hoff + kc * 32, EMB);

  v8f oacc[2][4];
#pragma unroll
  for (int mt = 0; mt < 2; ++mt)
#pragma unroll
    for (int dn = 0; dn < 4; ++dn)
      oacc[mt][dn] = (v8f){0.f,0.f,0.f,0.f,0.f,0.f,0.f,0.f};

  float mstat[2][8];   // running row max (uniform across the 16-lane half)
  float lsum[2][8];    // PER-LANE partial softmax denominator; lane-reduced once at the end
#pragma unroll
  for (int mt = 0; mt < 2; ++mt)
#pragma unroll
    for (int r = 0; r < 8; ++r) { mstat[mt][r] = -1e30f; lsum[mt][r] = 0.f; }

  for (int kb = 0; kb <= q0; kb += 64) {
    if (kb + 64 <= q0) {   // prefetch next K/V blocks (global_prefetch_b8)
      __builtin_prefetch(Kb + (size_t)(kb + 64 + lo16 * 4) * EMB + hoff, 0, 1);
      __builtin_prefetch(Vt + (size_t)(hoff + lo16 * 4) * SEQ + kb + 64, 0, 1);
    }

    // ---- S = Q * K^T for a 32x64 block (chained 16x16x32 bf16 WMMA over d_k) ----
    v8f s[2][4];
#pragma unroll
    for (int nt = 0; nt < 4; ++nt) {
      v16bf k0 = load_b_nt(Kb + (size_t)(kb + nt * 16) * EMB + hoff, EMB);
      v16bf k1 = load_b_nt(Kb + (size_t)(kb + nt * 16) * EMB + hoff + 32, EMB);
#pragma unroll
      for (int mt = 0; mt < 2; ++mt) {
        v8f t = (v8f){0.f,0.f,0.f,0.f,0.f,0.f,0.f,0.f};
        t = __builtin_amdgcn_wmma_f32_16x16x32_bf16(false, qa[mt][0], false, k0, (short)0, t, false, false);
        t = __builtin_amdgcn_wmma_f32_16x16x32_bf16(false, qa[mt][1], false, k1, (short)0, t, false, false);
        s[mt][nt] = t;
      }
    }

    // ---- causal mask (only blocks touching the diagonal) ----
    if (kb + 63 > q0) {
#pragma unroll
      for (int mt = 0; mt < 2; ++mt)
#pragma unroll
        for (int nt = 0; nt < 4; ++nt)
#pragma unroll
          for (int r = 0; r < 8; ++r) {
            const int qg = q0 + mt * 16 + r + 8 * hi;
            const int kg = kb + nt * 16 + lo16;
            if (kg > qg) s[mt][nt][r] = -1e30f;
          }
    }

    // ---- online softmax: only the row-max needs a cross-lane reduction ----
#pragma unroll
    for (int mt = 0; mt < 2; ++mt) {
#pragma unroll
      for (int r = 0; r < 8; ++r) {
        float t = fmaxf(fmaxf(s[mt][0][r], s[mt][1][r]), fmaxf(s[mt][2][r], s[mt][3][r]));
        t = fmaxf(t, __shfl_xor(t, 1, 32));
        t = fmaxf(t, __shfl_xor(t, 2, 32));
        t = fmaxf(t, __shfl_xor(t, 4, 32));
        t = fmaxf(t, __shfl_xor(t, 8, 32));
        const float mo = mstat[mt][r];
        const float mn = fmaxf(mo, t);
        const float sc = __expf(mo - mn);   // row-uniform -> per-lane partials rescale correctly
        mstat[mt][r] = mn;
        lsum[mt][r] *= sc;
#pragma unroll
        for (int dn = 0; dn < 4; ++dn) oacc[mt][dn][r] *= sc;
        float ps = 0.f;
#pragma unroll
        for (int nt = 0; nt < 4; ++nt) {
          const float p = __expf(s[mt][nt][r] - mn);
          s[mt][nt][r] = p;
          ps += p;
        }
        lsum[mt][r] += ps;                  // per-lane partial (this lane's 4 columns)
      }
    }

    // ---- P: C-layout -> A-operand layout via LDS ----
#pragma unroll
    for (int mt = 0; mt < 2; ++mt)
#pragma unroll
      for (int nt = 0; nt < 4; ++nt)
#pragma unroll
        for (int r = 0; r < 8; ++r)
          plds[wave][(mt * 16 + r + 8 * hi) * 64 + nt * 16 + lo16] = f2bf(s[mt][nt][r]);

    asm volatile("s_wait_dscnt 0" ::: "memory");

    v16bf pa[2][2];
#pragma unroll
    for (int mt = 0; mt < 2; ++mt)
#pragma unroll
      for (int kc = 0; kc < 2; ++kc) {
        const bf16_t* lp = &plds[wave][(mt * 16 + lo16) * 64 + kc * 32 + (hi << 3)];
        v8bf l0 = *(const v8bf*)(lp);
        v8bf l1 = *(const v8bf*)(lp + 16);
        pa[mt][kc] = __builtin_shufflevector(l0, l1, 0,1,2,3,4,5,6,7,8,9,10,11,12,13,14,15);
      }

    // ---- O += P * V  (Vt is [E][S], NT layout; K=64 -> 2 chained WMMAs) ----
#pragma unroll
    for (int dn = 0; dn < 4; ++dn) {
      v16bf v0 = load_b_nt(Vt + (size_t)(hoff + dn * 16) * SEQ + kb, SEQ);
      v16bf v1 = load_b_nt(Vt + (size_t)(hoff + dn * 16) * SEQ + kb + 32, SEQ);
#pragma unroll
      for (int mt = 0; mt < 2; ++mt) {
        oacc[mt][dn] = __builtin_amdgcn_wmma_f32_16x16x32_bf16(
            false, pa[mt][0], false, v0, (short)0, oacc[mt][dn], false, false);
        oacc[mt][dn] = __builtin_amdgcn_wmma_f32_16x16x32_bf16(
            false, pa[mt][1], false, v1, (short)0, oacc[mt][dn], false, false);
      }
    }
  }

  // ---- one final lane-reduction of the denominator, then normalize and store ----
#pragma unroll
  for (int mt = 0; mt < 2; ++mt) {
#pragma unroll
    for (int r = 0; r < 8; ++r) {
      float v = lsum[mt][r];
      v += __shfl_xor(v, 1, 32);
      v += __shfl_xor(v, 2, 32);
      v += __shfl_xor(v, 4, 32);
      v += __shfl_xor(v, 8, 32);
      const float inv = 1.0f / v;
      const int m = q0 + mt * 16 + r + 8 * hi;
#pragma unroll
      for (int dn = 0; dn < 4; ++dn)
        ctx[(size_t)m * EMB + hoff + dn * 16 + lo16] = f2bf(oacc[mt][dn][r] * inv);
    }
  }
}

extern "C" void kernel_launch(void* const* d_in, const int* in_sizes, int n_in,
                              void* d_out, int out_size, void* d_ws, size_t ws_size,
                              hipStream_t stream) {
  (void)in_sizes; (void)n_in; (void)out_size; (void)ws_size;
  const float* x  = (const float*)d_in[0];
  const float* Wq = (const float*)d_in[1];
  const float* bq = (const float*)d_in[2];
  const float* Wk = (const float*)d_in[3];
  const float* bk = (const float*)d_in[4];
  const float* Wv = (const float*)d_in[5];
  const float* bv = (const float*)d_in[6];
  const float* Wo = (const float*)d_in[7];
  const float* bo = (const float*)d_in[8];
  float* out = (float*)d_out;

  char* ws = (char*)d_ws;
  bf16_t* xb  = (bf16_t*)(ws);                                  // 8 MB
  bf16_t* wqb = (bf16_t*)(ws + (size_t)(8u << 20));             // 2 MB
  bf16_t* wkb = (bf16_t*)(ws + (size_t)(10u << 20));            // 2 MB
  bf16_t* wvb = (bf16_t*)(ws + (size_t)(12u << 20));            // 2 MB
  bf16_t* wob = (bf16_t*)(ws + (size_t)(14u << 20));            // 2 MB
  bf16_t* Qb  = (bf16_t*)(ws + (size_t)(16u << 20));            // 8 MB
  bf16_t* Kb  = (bf16_t*)(ws + (size_t)(24u << 20));            // 8 MB
  bf16_t* Vt  = (bf16_t*)(ws + (size_t)(32u << 20));            // 8 MB
  bf16_t* ctx = (bf16_t*)(ws + (size_t)(40u << 20));            // 8 MB (total 48 MB)

  cvt_f32_bf16<<<(SEQ * EMB + 255) / 256, 256, 0, stream>>>(x, xb, SEQ * EMB);
  cvt_f32_bf16<<<(EMB * EMB + 255) / 256, 256, 0, stream>>>(Wq, wqb, EMB * EMB);
  cvt_f32_bf16<<<(EMB * EMB + 255) / 256, 256, 0, stream>>>(Wk, wkb, EMB * EMB);
  cvt_f32_bf16<<<(EMB * EMB + 255) / 256, 256, 0, stream>>>(Wv, wvb, EMB * EMB);
  cvt_f32_bf16<<<(EMB * EMB + 255) / 256, 256, 0, stream>>>(Wo, wob, EMB * EMB);

  dim3 gg(EMB / 256, SEQ / 128);   // (4, 32)
  // Q gets the 1/sqrt(d_k) softmax scale folded in.
  gemm_bf16_wmma<1><<<gg, 256, 0, stream>>>(xb, wqb, bq, Qb, SEQ, EMB, EMB, 0.125f);
  gemm_bf16_wmma<1><<<gg, 256, 0, stream>>>(xb, wkb, bk, Kb, SEQ, EMB, EMB, 1.0f);
  gemm_bf16_wmma<2><<<gg, 256, 0, stream>>>(xb, wvb, bv, Vt, SEQ, EMB, EMB, 1.0f);

  attn_causal_wmma<<<(NH * (SEQ / 32)) / 4, 128, 0, stream>>>(Qb, Kb, Vt, ctx);

  gemm_bf16_wmma<0><<<gg, 256, 0, stream>>>(ctx, wob, bo, out, SEQ, EMB, EMB, 1.0f);
}
